// CausalPrefixAttention_43585328120008
// MI455X (gfx1250) — matrix operand order repaired and checked by reading, not compiled, not run on hardware
//
#include <hip/hip_runtime.h>

// ---------------- constants (match reference) ----------------
static constexpr int BB    = 2;
static constexpr int NN    = 1024;   // query length
static constexpr int CTXC  = 1024;   // context length
static constexpr int JJ    = CTXC + NN; // 2048 keys
static constexpr int DIMM  = 1024;
static constexpr int HH    = 16;
static constexpr int DH    = 64;
static constexpr int INNER = HH * DH; // 1024

typedef __attribute__((ext_vector_type(16))) _Float16 v16h;
typedef __attribute__((ext_vector_type(8)))  float    v8f;
typedef int v4i32 __attribute__((vector_size(16)));   // matches async builtin param

struct HF { union { v16h v; _Float16 h[16]; unsigned u[8]; uint4 q[2]; }; };

// CDNA5 async global->LDS (probe confirmed the builtin exists; param 0 is
// AS1 v4i32*, per the round-2 diagnostic)
#if __has_builtin(__builtin_amdgcn_global_load_async_to_lds_b128)
#define USE_ASYNC 1
#endif

__device__ __forceinline__ void async_wait0() {
#if __has_builtin(__builtin_amdgcn_s_wait_asynccnt)
  __builtin_amdgcn_s_wait_asynccnt(0);
#else
  asm volatile("s_wait_asynccnt 0" ::: "memory");
#endif
}

// interleave two rows of 8 halves into 8 u32 pairs (row-pair packing)
__device__ __forceinline__ void interleave_store(unsigned* dst, uint4 a, uint4 b) {
  unsigned pa[4] = {a.x, a.y, a.z, a.w};
  unsigned pb[4] = {b.x, b.y, b.z, b.w};
  unsigned o[8];
#pragma unroll
  for (int j = 0; j < 4; j++) {
    o[2 * j]     = (pa[j] & 0xffffu) | (pb[j] << 16);
    o[2 * j + 1] = (pa[j] >> 16)     | (pb[j] & 0xffff0000u);
  }
  *(uint4*)dst       = make_uint4(o[0], o[1], o[2], o[3]);
  *(uint4*)(dst + 4) = make_uint4(o[4], o[5], o[6], o[7]);
}

// =============================================================
// LayerNorm (x and context) -> f16, one block per row of 1024
// =============================================================
__global__ __launch_bounds__(256) void ln_f16_kernel(
    const float* __restrict__ x, const float* __restrict__ ctx,
    const float* __restrict__ nw, const float* __restrict__ nb,
    const float* __restrict__ cw, const float* __restrict__ cb,
    _Float16* __restrict__ xn, _Float16* __restrict__ cn)
{
  int row = blockIdx.x, tid = threadIdx.x;
  const float *src, *w, *bv; _Float16* dst;
  if (row < BB * NN) { src = x + (size_t)row * DIMM; dst = xn + (size_t)row * DIMM; w = nw; bv = nb; }
  else { int r = row - BB * NN; src = ctx + (size_t)r * DIMM; dst = cn + (size_t)r * DIMM; w = cw; bv = cb; }

  __shared__ float red[256];
  float vals[4]; float sum = 0.f;
#pragma unroll
  for (int i = 0; i < 4; i++) { vals[i] = src[tid + 256 * i]; sum += vals[i]; }
  red[tid] = sum; __syncthreads();
  for (int st = 128; st > 0; st >>= 1) { if (tid < st) red[tid] += red[tid + st]; __syncthreads(); }
  float mean = red[0] * (1.0f / DIMM);
  __syncthreads();
  float sq = 0.f;
#pragma unroll
  for (int i = 0; i < 4; i++) { float d = vals[i] - mean; sq += d * d; }
  red[tid] = sq; __syncthreads();
  for (int st = 128; st > 0; st >>= 1) { if (tid < st) red[tid] += red[tid + st]; __syncthreads(); }
  float var = red[0] * (1.0f / DIMM);
  float rstd = rsqrtf(var + 1e-5f);
#pragma unroll
  for (int i = 0; i < 4; i++) {
    int idx = tid + 256 * i;
    dst[idx] = (_Float16)((vals[i] - mean) * rstd * w[idx] + bv[idx]);
  }
}

// =============================================================
// fp32 -> f16 convert (weights)
// =============================================================
__global__ __launch_bounds__(256) void cvt_f16_kernel(
    const float* __restrict__ src, _Float16* __restrict__ dst, int n)
{
  int i = (blockIdx.x * 256 + threadIdx.x) * 4;
  if (i < n) {
    float4 v = *(const float4*)(src + i);
    dst[i + 0] = (_Float16)v.x; dst[i + 1] = (_Float16)v.y;
    dst[i + 2] = (_Float16)v.z; dst[i + 3] = (_Float16)v.w;
  }
}

// =============================================================
// Generic WMMA GEMM:  C[M,N] = A[M,K] * W[K,N]  (+bias)
// block tile 128x64, 8 waves of 32x32, K staged 32 wide in LDS.
// A tile row-major; W tile stored pair-interleaved [k/2][col] u32.
// =============================================================
__global__ __launch_bounds__(256) void gemm_f16_kernel(
    const _Float16* __restrict__ A, const _Float16* __restrict__ W,
    float* __restrict__ Cmat, int Mdim, int Kdim, int Ndim,
    const float* __restrict__ bias)
{
  (void)Mdim;
  __shared__ __align__(16) _Float16 lA[128 * 32];
  __shared__ __align__(16) unsigned lBp[16 * 64];   // u32 = (W[2p][c], W[2p+1][c])

  int tid = threadIdx.x;
  int wave = tid >> 5, lane = tid & 31, hi = lane >> 4, ln = lane & 15;
  int wm = wave & 3, wn = wave >> 2;               // wave -> 32x32 sub-tile
  int m0 = blockIdx.y * 128, n0 = blockIdx.x * 64;

  v8f zz = {};
  v8f acc[2][2];
  acc[0][0] = zz; acc[0][1] = zz; acc[1][0] = zz; acc[1][1] = zz;

  for (int kb = 0; kb < Kdim; kb += 32) {
    __syncthreads();
    // A tile 128x32 (512 uint4, 2 per thread)
#pragma unroll
    for (int i = 0; i < 2; i++) {
      int uu = tid * 2 + i; int rowA = uu >> 2, seg = uu & 3;
      *(uint4*)&lA[rowA * 32 + seg * 8] =
          *(const uint4*)(A + (size_t)(m0 + rowA) * Kdim + kb + seg * 8);
    }
    // W tile 32x64 pair-interleaved (128 tasks on waves 0-3)
    if (tid < 128) {
      int p = tid >> 3, c0 = (tid & 7) * 8;
      const _Float16* ra = W + (size_t)(kb + 2 * p) * Ndim + n0 + c0;
      uint4 a = *(const uint4*)ra;
      uint4 b = *(const uint4*)(ra + Ndim);
      interleave_store(&lBp[p * 64 + c0], a, b);
    }
    if (kb + 32 < Kdim) {
      __builtin_prefetch(A + (size_t)(m0 + (tid >> 1)) * Kdim + kb + 32);
      __builtin_prefetch(W + (size_t)(kb + 32 + (tid >> 3)) * Ndim + n0);
    }
    __syncthreads();

    HF af[2], bf[2];
#pragma unroll
    for (int tm = 0; tm < 2; tm++) {
      int mrow = wm * 32 + tm * 16 + ln;           // A row = lane&15
      af[tm].q[0] = *(const uint4*)&lA[mrow * 32 + 16 * hi];        // K 16hi..16hi+7
      af[tm].q[1] = *(const uint4*)&lA[mrow * 32 + 16 * hi + 8];   // K 16hi+8..16hi+15
    }
#pragma unroll
    for (int tn = 0; tn < 2; tn++) {
      int col = wn * 32 + tn * 16 + ln;            // B col = lane&15
#pragma unroll
      for (int vv = 0; vv < 8; vv++)
        bf[tn].u[vv] = lBp[(hi * 8 + vv) * 64 + col];  // k = hi*16+2v, 2v+1
    }
#pragma unroll
    for (int tm = 0; tm < 2; tm++)
#pragma unroll
      for (int tn = 0; tn < 2; tn++)
        acc[tm][tn] = __builtin_amdgcn_wmma_f32_16x16x32_f16(
            false, af[tm].v, false, bf[tn].v, (short)0, acc[tm][tn], false, false);
  }

#pragma unroll
  for (int tm = 0; tm < 2; tm++)
#pragma unroll
    for (int tn = 0; tn < 2; tn++)
#pragma unroll
      for (int r = 0; r < 8; r++) {
        int row = m0 + wm * 32 + tm * 16 + r + 8 * hi;   // C layout
        int col = n0 + wn * 32 + tn * 16 + ln;
        float v = acc[tm][tn][r];
        if (bias) v += bias[col];
        Cmat[(size_t)row * Ndim + col] = v;
      }
}

// =============================================================
// Build q/k/v head tensors [B,H,seq,64] f16 with rotary (+scale for q)
// =============================================================
__global__ __launch_bounds__(256) void build_qkv_kernel(
    int mode, const float* __restrict__ qproj, const float* __restrict__ kvx,
    const float* __restrict__ kvc, const float* __restrict__ rope,
    _Float16* __restrict__ dst)
{
  size_t i = (size_t)blockIdx.x * 256 + threadIdx.x;
  int seq = (mode == 0) ? NN : JJ;
  size_t total = (size_t)BB * HH * seq * DH;
  if (i >= total) return;
  int d = (int)(i & 63);
  size_t t1 = i >> 6;
  int s = (int)(t1 % seq); t1 /= seq;
  int h = (int)(t1 % HH); int b = (int)(t1 / HH);

  float outv;
  if (mode == 2) {
    int col = INNER + h * DH + d;
    outv = (s < CTXC) ? kvc[((size_t)b * CTXC + s) * (2 * INNER) + col]
                      : kvx[((size_t)b * NN + (s - CTXC)) * (2 * INNER) + col];
  } else {
    int dp = (d < 32) ? d + 32 : d - 32;
    const float* rowp;
    int p;
    if (mode == 0) {
      rowp = qproj + ((size_t)b * NN + s) * INNER + h * DH;
      p = (JJ - NN) + s;
    } else {
      rowp = (s < CTXC) ? (kvc + ((size_t)b * CTXC + s) * (2 * INNER) + h * DH)
                        : (kvx + ((size_t)b * NN + (s - CTXC)) * (2 * INNER) + h * DH);
      p = s;
    }
    float val = rowp[d], partner = rowp[dp];
    float th = rope[(size_t)p * DH + d];
    float cs = __cosf(th), sn = __sinf(th);
    outv = (d < 32) ? (val * cs - partner * sn) : (val * cs + partner * sn);
    if (mode == 0) outv *= 0.125f;   // 64^-0.5
  }
  dst[i] = (_Float16)outv;
}

// =============================================================
// Flash attention.  grid = (B*H, N/128); 8 waves x 16 query rows.
// K tile row-major in LDS (async copy when available);
// V tile pair-interleaved; P->A transpose through per-wave LDS.
// =============================================================
__global__ __launch_bounds__(256) void attn_kernel(
    const _Float16* __restrict__ qh, const _Float16* __restrict__ kh,
    const _Float16* __restrict__ vh, const int* __restrict__ cmask,
    _Float16* __restrict__ outh)
{
  constexpr int PRE = JJ - NN;   // 1024 causal offset
  int bh = blockIdx.x; int b = bh / HH; int h = bh % HH;
  int tid = threadIdx.x, wave = tid >> 5, lane = tid & 31, hi = lane >> 4, ln = lane & 15;
  int q0b = blockIdx.y * 128;
  int q0 = q0b + wave * 16;

  __shared__ __align__(16) _Float16 lK[32 * 64];
  __shared__ __align__(16) unsigned lVp[16 * 64];  // u32 = (V[2p][d], V[2p+1][d])
  __shared__ __align__(16) _Float16 lP[8][16 * 32];

  // Q fragments (16x64 row block, 2 chunks of K=32) as b128 loads
  HF qa[2];
  const _Float16* qrow = qh + ((size_t)bh * NN + q0 + ln) * DH;
#pragma unroll
  for (int c = 0; c < 2; c++) {
    qa[c].q[0] = *(const uint4*)(qrow + c * 32 + 16 * hi);
    qa[c].q[1] = *(const uint4*)(qrow + c * 32 + 16 * hi + 8);
  }

  v8f zz = {};
  v8f o[4]; o[0] = zz; o[1] = zz; o[2] = zz; o[3] = zz;
  float mrow[8], lrow[8];
#pragma unroll
  for (int r = 0; r < 8; r++) { mrow[r] = -1e30f; lrow[r] = 0.f; }

  int nj = q0b + 128 + PRE;            // block-uniform key bound (<= J)

  for (int j0 = 0; j0 < nj; j0 += 32) {
    __syncthreads();
    if (tid < 128) {
      // V tile pair-interleave (waves 0-3)
      int p = tid >> 3, c0 = (tid & 7) * 8;
      const _Float16* vr = vh + ((size_t)bh * JJ + j0 + 2 * p) * DH + c0;
      uint4 a = *(const uint4*)vr;
      uint4 b2 = *(const uint4*)(vr + DH);
      interleave_store(&lVp[p * 64 + c0], a, b2);
    } else {
      // K tile row-major copy (waves 4-7), async when available
      int task = tid - 128;
#pragma unroll
      for (int i = 0; i < 2; i++) {
        int uu = task * 2 + i; int row = uu >> 3, sg = uu & 7;
        const _Float16* src = kh + ((size_t)bh * JJ + j0 + row) * DH + sg * 8;
#if defined(USE_ASYNC)
        __builtin_amdgcn_global_load_async_to_lds_b128(
            (__attribute__((address_space(1))) v4i32*)(void*)src,
            (__attribute__((address_space(3))) v4i32*)(void*)&lK[row * 64 + sg * 8],
            0, 0);
#else
        *(uint4*)&lK[row * 64 + sg * 8] = *(const uint4*)src;
#endif
      }
    }
    if (j0 + 32 < nj)
      __builtin_prefetch(kh + ((size_t)bh * JJ + j0 + 32 + (tid >> 3)) * DH);
#if defined(USE_ASYNC)
    async_wait0();
#endif
    __syncthreads();

    // S = q . k^T  (two 16x16 col tiles, K=64 split 2x32)
    v8f s[2]; s[0] = zz; s[1] = zz;
#pragma unroll
    for (int t = 0; t < 2; t++)
#pragma unroll
      for (int c = 0; c < 2; c++) {
        HF kb;
        const _Float16* krow = &lK[(t * 16 + ln) * DH + c * 32 + hi * 16];
        kb.q[0] = *(const uint4*)krow;        // B(d,n)=K[n][d], d pairs contiguous
        kb.q[1] = *(const uint4*)(krow + 8);
        s[t] = __builtin_amdgcn_wmma_f32_16x16x32_f16(
            false, qa[c].v, false, kb.v, (short)0, s[t], false, false);
      }

    // online softmax (C layout: row = r + 8*hi, col = ln)
#pragma unroll
    for (int r = 0; r < 8; r++) {
      int row = q0 + r + 8 * hi;
      int c0 = j0 + ln, c1 = c0 + 16;
      bool ok0 = (c0 <= row + PRE) && (c0 >= CTXC || cmask[b * CTXC + c0] != 0);
      bool ok1 = (c1 <= row + PRE) && (c1 >= CTXC || cmask[b * CTXC + c1] != 0);
      float s0 = ok0 ? s[0][r] : -1e30f;
      float s1 = ok1 ? s[1][r] : -1e30f;
      float t = fmaxf(s0, s1);
#pragma unroll
      for (int ms = 1; ms < 16; ms <<= 1) t = fmaxf(t, __shfl_xor(t, ms, 32));
      float mn = fmaxf(mrow[r], t);
      float al = __expf(mrow[r] - mn);
      float p0 = __expf(s0 - mn), p1 = __expf(s1 - mn);
      float rs = p0 + p1;
#pragma unroll
      for (int ms = 1; ms < 16; ms <<= 1) rs += __shfl_xor(rs, ms, 32);
      lrow[r] = lrow[r] * al + rs;
      mrow[r] = mn;
      o[0][r] *= al; o[1][r] *= al; o[2][r] *= al; o[3][r] *= al;
      lP[wave][(r + 8 * hi) * 32 + ln]      = (_Float16)p0;
      lP[wave][(r + 8 * hi) * 32 + 16 + ln] = (_Float16)p1;
    }
    asm volatile("s_wait_dscnt 0" ::: "memory");   // wave-local P transpose RAW

    HF pa;
    pa.q[0] = *(const uint4*)&lP[wave][ln * 32 + 16 * hi];
    pa.q[1] = *(const uint4*)&lP[wave][ln * 32 + 16 * hi + 8];

    // O += P . V  (4 output column groups of 16)
#pragma unroll
    for (int g = 0; g < 4; g++) {
      HF vb;
#pragma unroll
      for (int vv = 0; vv < 8; vv++)
        vb.u[vv] = lVp[(hi * 8 + vv) * 64 + g * 16 + ln];  // k = hi*16+2v, 2v+1
      o[g] = __builtin_amdgcn_wmma_f32_16x16x32_f16(
          false, pa.v, false, vb.v, (short)0, o[g], false, false);
    }
  }

#pragma unroll
  for (int r = 0; r < 8; r++) {
    int row = q0 + r + 8 * hi;
    float inv = 1.0f / lrow[r];
#pragma unroll
    for (int g = 0; g < 4; g++)
      outh[((size_t)b * NN + row) * INNER + h * DH + g * 16 + ln] = (_Float16)(o[g][r] * inv);
  }
}

// =============================================================
// host launch
// =============================================================
extern "C" void kernel_launch(void* const* d_in, const int* in_sizes, int n_in,
                              void* d_out, int out_size, void* d_ws, size_t ws_size,
                              hipStream_t stream) {
  (void)in_sizes; (void)n_in; (void)out_size; (void)ws_size;
  const float* x       = (const float*)d_in[0];
  const float* context = (const float*)d_in[1];
  const int*   cmask   = (const int*)d_in[2];
  const float* rope    = (const float*)d_in[3];
  const float* nw      = (const float*)d_in[4];
  const float* nb      = (const float*)d_in[5];
  const float* cw      = (const float*)d_in[6];
  const float* cb      = (const float*)d_in[7];
  const float* Wq      = (const float*)d_in[8];
  const float* Wkv     = (const float*)d_in[9];
  const float* Wo      = (const float*)d_in[10];
  const float* bo      = (const float*)d_in[11];
  float* out = (float*)d_out;

  char* base = (char*)d_ws;
  size_t off = 0;
  auto take = [&](size_t bytes) -> char* {
    char* p = base + off;
    off += bytes; off = (off + 255) & ~(size_t)255;
    return p;
  };
  _Float16* xn_h  = (_Float16*)take((size_t)BB * NN * DIMM * 2);
  _Float16* cn_h  = (_Float16*)take((size_t)BB * CTXC * DIMM * 2);
  _Float16* Wq_h  = (_Float16*)take((size_t)DIMM * INNER * 2);
  _Float16* Wkv_h = (_Float16*)take((size_t)DIMM * 2 * INNER * 2);
  _Float16* Wo_h  = (_Float16*)take((size_t)INNER * DIMM * 2);
  float*    qproj = (float*)take((size_t)BB * NN * INNER * 4);
  float*    kvx   = (float*)take((size_t)BB * NN * 2 * INNER * 4);
  float*    kvc   = (float*)take((size_t)BB * CTXC * 2 * INNER * 4);
  _Float16* q_h   = (_Float16*)take((size_t)BB * HH * NN * DH * 2);
  _Float16* k_h   = (_Float16*)take((size_t)BB * HH * JJ * DH * 2);
  _Float16* v_h   = (_Float16*)take((size_t)BB * HH * JJ * DH * 2);
  _Float16* out_h = (_Float16*)take((size_t)BB * NN * INNER * 2);

  dim3 blk(256);

  ln_f16_kernel<<<BB * NN + BB * CTXC, blk, 0, stream>>>(
      x, context, nw, nb, cw, cb, xn_h, cn_h);

  cvt_f16_kernel<<<(DIMM * INNER) / 1024, blk, 0, stream>>>(Wq, Wq_h, DIMM * INNER);
  cvt_f16_kernel<<<(DIMM * 2 * INNER) / 1024, blk, 0, stream>>>(Wkv, Wkv_h, DIMM * 2 * INNER);
  cvt_f16_kernel<<<(INNER * DIMM) / 1024, blk, 0, stream>>>(Wo, Wo_h, INNER * DIMM);

  gemm_f16_kernel<<<dim3(INNER / 64, (BB * NN) / 128), blk, 0, stream>>>(
      xn_h, Wq_h, qproj, BB * NN, DIMM, INNER, nullptr);
  gemm_f16_kernel<<<dim3((2 * INNER) / 64, (BB * NN) / 128), blk, 0, stream>>>(
      xn_h, Wkv_h, kvx, BB * NN, DIMM, 2 * INNER, nullptr);
  gemm_f16_kernel<<<dim3((2 * INNER) / 64, (BB * CTXC) / 128), blk, 0, stream>>>(
      cn_h, Wkv_h, kvc, BB * CTXC, DIMM, 2 * INNER, nullptr);

  build_qkv_kernel<<<(BB * HH * NN * DH) / 256, blk, 0, stream>>>(0, qproj, kvx, kvc, rope, q_h);
  build_qkv_kernel<<<(BB * HH * JJ * DH) / 256, blk, 0, stream>>>(1, qproj, kvx, kvc, rope, k_h);
  build_qkv_kernel<<<(BB * HH * JJ * DH) / 256, blk, 0, stream>>>(2, qproj, kvx, kvc, rope, v_h);

  attn_kernel<<<dim3(BB * HH, NN / 128), blk, 0, stream>>>(q_h, k_h, v_h, cmask, out_h);

  gemm_f16_kernel<<<dim3(DIMM / 64, (BB * NN) / 128), blk, 0, stream>>>(
      out_h, Wo_h, out, BB * NN, INNER, DIMM, bo);
}
